// AdaptiveFullConnected_48670569398908
// MI455X (gfx1250) — matrix operand order repaired and checked by reading, not compiled, not run on hardware
//
#include <hip/hip_runtime.h>
#include <hip/hip_bf16.h>
#include <math.h>

// Problem constants (match reference)
#define BATCH  4
#define SEQN   16384
#define DIMSZ  64
#define HEADS  8
#define DMODEL 512
#define PSEG   256
#define KW     5

// Tiling
#define BM      64          // rows (of B*N) per workgroup
#define HSTRIDE 260         // dwords per LDS row of h (256 + 4 pad): 16B aligned, conflict-free
#define BSTRIDE 20          // dwords per LDS row of staged B slice (16 + 4 TDM pad)

typedef __attribute__((ext_vector_type(16))) __bf16       v16bf;
typedef __attribute__((ext_vector_type(8)))  float        v8f;
typedef __attribute__((ext_vector_type(4)))  unsigned int v4u;
typedef __attribute__((ext_vector_type(4)))  unsigned int u32x4;
typedef __attribute__((ext_vector_type(8)))  int          i32x8;
typedef __attribute__((ext_vector_type(4)))  int          i32x4;

union Frag {
    v16bf bf;
    v4u   q[2];
};

__device__ __forceinline__ unsigned int pack_bf16(float a, float b) {
    unsigned int ua = __float_as_uint(a);
    unsigned int ub = __float_as_uint(b);
    ua = (ua + 0x7FFFu + ((ua >> 16) & 1u)) >> 16;   // RNE f32 -> bf16
    ub = (ub + 0x7FFFu + ((ub >> 16) & 1u)) >> 16;
    return (ub << 16) | (ua & 0xFFFFu);
}

__device__ __forceinline__ float gelu_exact(float v) {
    return 0.5f * v * (1.0f + erff(v * 0.70710678118654752f));
}

// TDM: load one K-slice of packed B (2D tile: 16 dwords wide x 512 rows,
// row stride 256 dwords in memory) into LDS with hardware padding
// (pad_interval=3 -> every 16 dwords, pad_amount=3 -> +4 dwords) giving an
// LDS row stride of 20 dwords (bank-conflict-free fragment reads).
__device__ __forceinline__ void tdm_load_slice(const unsigned int* gsrc,
                                               unsigned int* ldsdst) {
    unsigned long long ga = (unsigned long long)(uintptr_t)gsrc;
    unsigned int lds_addr = (unsigned int)(uintptr_t)ldsdst;  // low 32 bits = LDS byte offset

    u32x4 g0;
    g0[0] = 1u;                                              // count=1, user load
    g0[1] = lds_addr;                                        // lds_addr
    g0[2] = (unsigned int)(ga & 0xFFFFFFFFull);              // global_addr[31:0]
    g0[3] = (unsigned int)((ga >> 32) & 0x01FFFFFFull)       // global_addr[56:32]
          | (2u << 30);                                      // type = 2 ("image")

    i32x8 g1;
    g1[0] = (int)((2u << 16)    // data_size = 4 bytes
                | (1u << 20)    // pad_enable
                | (3u << 22)    // pad_interval: every 16 dwords
                | (3u << 25));  // pad_amount: 4 dwords
    g1[1] = (int)(16u << 16);   // tensor_dim0 = 16 (atomic_barrier_addr = 0)
    g1[2] = (int)(512u << 16);  // tensor_dim0 hi = 0 ; tensor_dim1 = 512
    g1[3] = (int)(16u << 16);   // tensor_dim1 hi = 0 ; tile_dim0 = 16
    g1[4] = (int)(512u);        // tile_dim1 = 512 ; tile_dim2 = 0
    g1[5] = (int)(256u);        // tensor_dim0_stride = 256 dwords
    g1[6] = 0;
    g1[7] = 0;

    i32x4 gz4 = {0, 0, 0, 0};               // groups 2/3: 2-D tensor, all zero
    i32x8 gz8 = {0, 0, 0, 0, 0, 0, 0, 0};   // 6-arg toolchain: extra group
    __builtin_amdgcn_tensor_load_to_lds(g0, g1, gz4, gz4, gz8, 0);
}

// ---------------------------------------------------------------------------
// Prep kernels
// ---------------------------------------------------------------------------
__global__ void prep_zero(float* p, int n) {
    int i = blockIdx.x * blockDim.x + threadIdx.x;
    if (i < n) p[i] = 0.0f;
}

// lin2_w [512][512] f32 -> packed bf16 pairs W[512][256] (d-major, k contiguous)
__global__ void prep_pack_w(const float* __restrict__ lin2_w,
                            unsigned int* __restrict__ W) {
    int i = blockIdx.x * blockDim.x + threadIdx.x;   // 131072
    int d  = i >> 8;
    int k2 = i & 255;
    const float* p = lin2_w + d * DMODEL + 2 * k2;
    W[i] = pack_bf16(p[0], p[1]);
}

__global__ void prep_counts(const int* __restrict__ indices,
                            float* __restrict__ counts) {
    int gm = blockIdx.x * blockDim.x + threadIdx.x;  // B*N = 65536
    int b  = gm >> 14;                               // / SEQN
    int s  = indices[gm];
    atomicAdd(counts + b * PSEG + s, 1.0f);
}

// ---------------------------------------------------------------------------
// Fused: conv1d + lin1(gelu) -> bf16 LDS tile -> WMMA GEMM (B staged by TDM)
//        -> * tiled(x) -> segment-sum atomics
// One workgroup = 64 rows of the flattened [B*N] dimension.
// 8 waves: wave w -> M-tile (w>>1) of 16 rows, output-dim half (w&1) of 256.
// ---------------------------------------------------------------------------
__global__ __launch_bounds__(256)
void fused_gemm(const float* __restrict__ x,
                const float* __restrict__ coords,
                const int*   __restrict__ indices,
                const float* __restrict__ conv_w,
                const float* __restrict__ conv_b,
                const float* __restrict__ lin1_w,
                const float* __restrict__ lin1_b,
                const unsigned int* __restrict__ Wb,   // packed bf16 lin2_w [512][256]
                const float* __restrict__ lin2_b,
                float* __restrict__ sums)
{
    __shared__ unsigned int hlds[BM * HSTRIDE];         // h tile, bf16 pairs (~65KB)
    __shared__ unsigned int Blds[2][DMODEL * BSTRIDE];  // double-buffered B K-slices (2x40KB)
    __shared__ float c0a[BM], c1a[BM];                  // conv outputs per row
    __shared__ float cs0[BM + 4], cs1[BM + 4];          // coords with halo

    const int tid  = threadIdx.x;
    const int wave = tid >> 5;
    const int m0   = blockIdx.x * BM;             // first flat row
    const int b0   = m0 >> 14;                    // batch (BM divides SEQN)
    const int n0   = m0 & (SEQN - 1);

    // ---- kick off TDM preloads of B slices 0 and 1 (overlap with h phase) ----
    if (wave == 0) {
        tdm_load_slice(Wb + 0,  &Blds[0][0]);
        tdm_load_slice(Wb + 16, &Blds[1][0]);
    }

    // ---- stage coords (zero-padded halo of 2 on each side, per batch) ----
    for (int i = tid; i < BM + 4; i += 256) {
        int n = n0 + i - 2;
        bool ok = (n >= 0) && (n < SEQN);
        int ci = (b0 * SEQN + (ok ? n : 0)) * 2;
        cs0[i] = ok ? coords[ci + 0] : 0.0f;
        cs1[i] = ok ? coords[ci + 1] : 0.0f;
    }
    __syncthreads();

    // ---- depthwise conv (K=5, cross-correlation) + residual + bias ----
    if (tid < BM) {
        float a0 = 0.0f, a1 = 0.0f;
        #pragma unroll
        for (int j = 0; j < KW; ++j) {
            a0 += conv_w[j]      * cs0[tid + j];
            a1 += conv_w[KW + j] * cs1[tid + j];
        }
        c0a[tid] = cs0[tid + 2] + a0 + conv_b[0];
        c1a[tid] = cs1[tid + 2] + a1 + conv_b[1];
    }
    __syncthreads();

    // ---- h = gelu(c @ lin1_w^T + lin1_b), pack to bf16 in LDS ----
    for (int idx = tid; idx < BM * 256; idx += 256) {
        int r = idx >> 8;          // row in tile
        int p = idx & 255;         // bf16 pair index (cols 2p, 2p+1)
        float4 L  = *(const float4*)(lin1_w + 4 * p);   // [2p][0..1],[2p+1][0..1]
        float2 Bv = *(const float2*)(lin1_b + 2 * p);
        float cr0 = c0a[r], cr1 = c1a[r];
        float h0 = gelu_exact(cr0 * L.x + cr1 * L.y + Bv.x);
        float h1 = gelu_exact(cr0 * L.z + cr1 * L.w + Bv.y);
        hlds[r * HSTRIDE + p] = pack_bf16(h0, h1);
    }

    // slice 0 must be resident before the first MAC iteration
    if (wave == 0) __builtin_amdgcn_s_wait_tensorcnt(1);
    __syncthreads();

    // ---- WMMA GEMM: weight[BM][512] = h[BM][512] @ lin2_w^T ----
    const int lane   = tid & 31;
    const int m_tile = wave >> 1;      // 0..3
    const int n_half = wave & 1;       // 0..1 (which 256 output dims)
    const int lhi    = lane >> 4;      // 0/1
    const int llo    = lane & 15;

    v8f acc[16] = {};                  // 16 N-tiles x 16x16 f32

    // A fragment: lane reads row (m_tile*16 + llo); pair offsets (contiguous k, matches B):
    //   v0..3 -> pairs (lhi*8 + 0..3), v4..7 -> pairs (lhi*8 + 4..7), + K0/2
    const unsigned int* arow = hlds + (m_tile * 16 + llo) * HSTRIDE + lhi * 8;
    // B fragment from staged slice: row d = n_half*256 + t*16 + llo,
    //   dwords lhi*8 + v (v = 0..7 contiguous), LDS row stride 20 dwords
    const int brow0 = (n_half * 256 + llo) * BSTRIDE + lhi * 8;

    #pragma unroll 1
    for (int it = 0; it < 16; ++it) {
        const int K0  = it * 32;
        const int buf = it & 1;

        Frag a;
        a.q[0] = *(const v4u*)(arow + (K0 >> 1));
        a.q[1] = *(const v4u*)(arow + (K0 >> 1) + 4);

        const unsigned int* bbase = &Blds[buf][0] + brow0;

        // 4-deep software pipeline with a 5-buffer rotation, pinned by
        // sched_group_barrier so pre-RA scheduling keeps the fragment loads
        // 4 tiles ahead of their WMMA (waits become dscnt<=6, not drains).
        Frag bb[5];
        #pragma unroll
        for (int pt = 0; pt < 4; ++pt) {
            const unsigned int* bp = bbase + pt * 16 * BSTRIDE;
            bb[pt].q[0] = *(const v4u*)(bp);
            bb[pt].q[1] = *(const v4u*)(bp + 4);
        }
        #pragma unroll
        for (int t = 0; t < 16; ++t) {
            if (t + 4 < 16) {
                const unsigned int* bp = bbase + (t + 4) * 16 * BSTRIDE;
                bb[(t + 4) % 5].q[0] = *(const v4u*)(bp);
                bb[(t + 4) % 5].q[1] = *(const v4u*)(bp + 4);
            }
            acc[t] = __builtin_amdgcn_wmma_f32_16x16x32_bf16(
                false, a.bf, false, bb[t % 5].bf, (short)0, acc[t], false, false);
        }
        // Pipeline shape for this region:
        //   10 DS-reads (A frag + B tiles 0..3), then 12x(1 WMMA, 2 DS-reads),
        //   then the last 4 WMMAs.  0x100 = DS-read group, 0x8 = MFMA/WMMA.
        __builtin_amdgcn_sched_group_barrier(0x100, 10, 0);
        #pragma unroll
        for (int t = 0; t < 12; ++t) {
            __builtin_amdgcn_sched_group_barrier(0x008, 1, 0);
            __builtin_amdgcn_sched_group_barrier(0x100, 2, 0);
        }
        __builtin_amdgcn_sched_group_barrier(0x008, 4, 0);

        if (it < 15) {
            __syncthreads();                 // everyone done reading `buf`
            if (wave == 0) {
                if (it + 2 < 16) {           // refill `buf` with slice it+2
                    tdm_load_slice(Wb + (it + 2) * 16, &Blds[buf][0]);
                    // TDM completes in order: <=1 outstanding => slice it+1 landed
                    __builtin_amdgcn_s_wait_tensorcnt(1);
                } else {
                    __builtin_amdgcn_s_wait_tensorcnt(0);
                }
            }
            __syncthreads();                 // slice it+1 visible to all waves
        }
    }

    // ---- epilogue: + lin2_b, * tiled(x), segment-sum atomics ----
    // C layout: VGPR r, lanes 0-15 -> M=r, lanes 16-31 -> M=8+r; N = llo
    const int mbase = m0 + m_tile * 16 + lhi * 8;
    int soff[8], xoff[8];
    #pragma unroll
    for (int r = 0; r < 8; ++r) {
        int gm  = mbase + r;
        int bb2 = gm >> 14;
        int seg = indices[gm];
        soff[r] = (bb2 * PSEG + seg) * DMODEL;
        xoff[r] = gm * DIMSZ;
    }
    #pragma unroll
    for (int t = 0; t < 16; ++t) {
        int d    = n_half * 256 + t * 16 + llo;
        float bi = lin2_b[d];
        int dm   = d & (DIMSZ - 1);
        #pragma unroll
        for (int r = 0; r < 8; ++r) {
            float val = acc[t][r] + bi;
            float xv  = x[xoff[r] + dm];
            atomicAdd(sums + soff[r] + d, val * xv);
        }
    }
}

// ---------------------------------------------------------------------------
// Finalize: mean = sums/counts; out = w1*sin(mean) + w2*cos(mean)
// ---------------------------------------------------------------------------
__global__ void finalize(const float* __restrict__ sums,
                         const float* __restrict__ counts,
                         const float* __restrict__ w1,
                         const float* __restrict__ w2,
                         float* __restrict__ out) {
    int i  = blockIdx.x * blockDim.x + threadIdx.x;  // B*P*D = 524288
    int bp = i >> 9;                                 // / DMODEL
    float m = sums[i] / counts[bp];
    out[i] = w1[0] * sinf(m) + w2[0] * cosf(m);
}

// ---------------------------------------------------------------------------
extern "C" void kernel_launch(void* const* d_in, const int* in_sizes, int n_in,
                              void* d_out, int out_size, void* d_ws, size_t ws_size,
                              hipStream_t stream) {
    (void)in_sizes; (void)n_in; (void)out_size; (void)ws_size;

    const float* x       = (const float*)d_in[0];
    const float* coords  = (const float*)d_in[1];
    const int*   indices = (const int*)d_in[2];
    // d_in[3] = patch_seq_len (== 256, compiled in)
    const float* conv_w  = (const float*)d_in[4];
    const float* conv_b  = (const float*)d_in[5];
    const float* lin1_w  = (const float*)d_in[6];
    const float* lin1_b  = (const float*)d_in[7];
    const float* lin2_w  = (const float*)d_in[8];
    const float* lin2_b  = (const float*)d_in[9];
    const float* w1      = (const float*)d_in[10];
    const float* w2      = (const float*)d_in[11];

    // Workspace layout:
    //   [0, 2MB)          : sums   float[B][P][D]
    //   [2MB, 2MB+4KB)    : counts float[B][P]
    //   [2MB+4KB, +512KB) : packed bf16 lin2_w  uint32[512][256]
    char* ws = (char*)d_ws;
    float*        sums   = (float*)ws;
    float*        counts = (float*)(ws + 2u * 1024u * 1024u);
    unsigned int* Wb     = (unsigned int*)(ws + 2u * 1024u * 1024u + 4096u);

    const int nz = BATCH * PSEG * DMODEL + BATCH * PSEG;  // sums+counts contiguous
    prep_zero   <<<(nz + 255) / 256, 256, 0, stream>>>(sums, nz);
    prep_pack_w <<<(DMODEL * DMODEL / 2) / 256, 256, 0, stream>>>(lin2_w, Wb);
    prep_counts <<<(BATCH * SEQN) / 256, 256, 0, stream>>>(indices, counts);

    fused_gemm  <<<(BATCH * SEQN) / BM, 256, 0, stream>>>(
        x, coords, indices, conv_w, conv_b, lin1_w, lin1_b, Wb, lin2_b, sums);

    finalize    <<<(BATCH * PSEG * DMODEL) / 256, 256, 0, stream>>>(
        sums, counts, w1, w2, (float*)d_out);
}
